// MSGEncoderLSTM_51367808860616
// MI455X (gfx1250) — matrix operand order repaired and checked by reading, not compiled, not run on hardware
//
#include <hip/hip_runtime.h>
#include <hip/hip_bf16.h>

// ---------------------------------------------------------------------------
// LSTM (torch LSTMCell gate order i,f,g,o) on gfx1250 via bf16 WMMA, fp32 acc.
// T=128, B=512, I=64, H=1024.
// Persistent kernel: all 128 serially-dependent timesteps run inside one
// kernel; steps are separated by a device-wide software barrier (atomic
// counter + s_sleep spin + agent-scope release/acquire fences) instead of 128
// kernel launches. Grid is sized to min(128, #processors) and each block
// strides over the 128 (m-superblock x n-tile) work slots, so all launched
// blocks are resident -> no deadlock.
// Per slot: block = 8 waves owns one n-tile (16 cols x 4 gates) and 256 batch
// rows; W_hh fragments staged through LDS (double-buffered, async-to-LDS DMA
// when available), each wave register-blocks 2 M-tiles over the shared B.
// ---------------------------------------------------------------------------

#define TT 128
#define BB 512
#define II 64
#define HH 1024
#define G4H (4 * HH)
#define KC  128              // K-chunk staged in LDS
#define LDSPITCH 136         // 128 + 8 pad (272 B row stride -> bank spread)
#define NSLOTS 128           // 2 m-superblocks x 64 n-tiles

typedef __attribute__((ext_vector_type(16))) __bf16 v16bf;
typedef __attribute__((ext_vector_type(8)))  float  v8f;
typedef __attribute__((ext_vector_type(4)))  unsigned int u32x4;

// gfx1250 async global->LDS DMA (ASYNCcnt) if the toolchain exposes it.
#if defined(__has_builtin)
#if __has_builtin(__builtin_amdgcn_global_load_async_to_lds_b128) && \
    __has_builtin(__builtin_amdgcn_s_wait_asynccnt)
#define HAVE_ASYNC_LDS 1
#endif
#endif
#ifndef HAVE_ASYNC_LDS
#define HAVE_ASYNC_LDS 0
#endif

// The builtin's pointee type is 'int __attribute__((vector_size(16)))'
// (from the compiler diagnostic); param 1 is the global (AS1) source,
// param 2 the LDS (AS3) destination.
typedef int i32x4v __attribute__((vector_size(16)));
typedef __attribute__((address_space(1))) i32x4v gvec_t;
typedef __attribute__((address_space(3))) i32x4v lvec_t;
#define TO_GLOBAL(p) ((gvec_t*)(size_t)(p))
#define TO_LDS(p)    ((lvec_t*)(unsigned int)(size_t)(p))   // low 32b = LDS offset

union FragU {
    u32x4 q[2];   // two 16-byte chunks = 32 bytes = 16 bf16
    v16bf v;
};

__device__ __forceinline__ unsigned short f2bf_rne(float f) {
    unsigned int u = __float_as_uint(f);
    u += 0x7FFFu + ((u >> 16) & 1u);   // round-to-nearest-even
    return (unsigned short)(u >> 16);
}

__device__ __forceinline__ float sigmoid_f(float x) {
    return 1.0f / (1.0f + __expf(-x));
}

__device__ __forceinline__ float tanh_f(float x) {
    // 1 - 2/(e^{2x}+1): exact limits at +/-inf, no inf/inf NaN.
    return 1.0f - 2.0f / (__expf(2.0f * x) + 1.0f);
}

__device__ __forceinline__ v8f wmma_bf16(const FragU& a, const FragU& b, v8f c) {
    return __builtin_amdgcn_wmma_f32_16x16x32_bf16(
        false, a.v, false, b.v, (short)0, c, false, false);
}

// ---------------------------------------------------------------------------
// Pack: fp32 -> bf16 for W_ih, W_hh, X; bias = b_ih + b_hh; init h, c, hbf0;
// reset the device-wide barrier counter (replay-safe).
// Launched with exactly 4,194,304 threads (largest segment).
// ---------------------------------------------------------------------------
__global__ void lstm_pack_kernel(const float* __restrict__ X,
                                 const float* __restrict__ Wih,
                                 const float* __restrict__ Whh,
                                 const float* __restrict__ bih,
                                 const float* __restrict__ bhh,
                                 const float* __restrict__ h0,
                                 const float* __restrict__ c0,
                                 unsigned short* __restrict__ Xbf,
                                 unsigned short* __restrict__ Wihbf,
                                 unsigned short* __restrict__ Whhbf,
                                 float* __restrict__ bias,
                                 float* __restrict__ h,
                                 float* __restrict__ c,
                                 unsigned short* __restrict__ hbf0,
                                 unsigned int* __restrict__ bar) {
    const unsigned int i = blockIdx.x * blockDim.x + threadIdx.x;
    // W_hh: 4096*1024, X: 128*512*64 (both 4,194,304)
    Whhbf[i] = f2bf_rne(Whh[i]);
    Xbf[i]   = f2bf_rne(X[i]);
    if (i < (unsigned)(G4H * II)) {               // 262,144
        Wihbf[i] = f2bf_rne(Wih[i]);
    }
    if (i < (unsigned)G4H) {                      // 4,096
        bias[i] = bih[i] + bhh[i];
    }
    if (i < (unsigned)(BB * HH)) {                // 524,288
        const float hv = h0[i & (HH - 1)];
        const float cv = c0[i & (HH - 1)];
        h[i]    = hv;
        c[i]    = cv;
        hbf0[i] = f2bf_rne(hv);
    }
    if (i == 0) { *bar = 0u; }
}

// ---------------------------------------------------------------------------
// Cooperative staging: block fetches one K-chunk of B (4 gates x 16 rows x
// 128 k = 16 KB) -- 64 B per thread (4 x b128).
// ---------------------------------------------------------------------------
#if HAVE_ASYNC_LDS
__device__ __forceinline__ void stage_async(const unsigned short* __restrict__ W,
                                            int n0, int kc, int tid,
                                            unsigned short* __restrict__ buf) {
    const int row = tid >> 2;                 // 0..63 : gate = row>>4, col = row&15
    const int q   = row >> 4;
    const int rc  = row & 15;
    const unsigned short* gp =
        W + (size_t)(q * HH + n0 + rc) * HH + kc + (tid & 3) * 32;
    unsigned short* lp = buf + row * LDSPITCH + (tid & 3) * 32;
    __builtin_amdgcn_global_load_async_to_lds_b128(TO_GLOBAL(gp),      TO_LDS(lp),      0, 0);
    __builtin_amdgcn_global_load_async_to_lds_b128(TO_GLOBAL(gp + 8),  TO_LDS(lp + 8),  0, 0);
    __builtin_amdgcn_global_load_async_to_lds_b128(TO_GLOBAL(gp + 16), TO_LDS(lp + 16), 0, 0);
    __builtin_amdgcn_global_load_async_to_lds_b128(TO_GLOBAL(gp + 24), TO_LDS(lp + 24), 0, 0);
}
#else
__device__ __forceinline__ void stage_fetch(const unsigned short* __restrict__ W,
                                            int n0, int kc, int tid, u32x4 reg[4]) {
    const int row = tid >> 2;
    const int q   = row >> 4;
    const int rc  = row & 15;
    const unsigned short* p =
        W + (size_t)(q * HH + n0 + rc) * HH + kc + (tid & 3) * 32;
    reg[0] = *(const u32x4*)(p);
    reg[1] = *(const u32x4*)(p + 8);
    reg[2] = *(const u32x4*)(p + 16);
    reg[3] = *(const u32x4*)(p + 24);
}

__device__ __forceinline__ void stage_store(unsigned short* __restrict__ buf,
                                            int tid, const u32x4 reg[4]) {
    const int row = tid >> 2;
    unsigned short* p = buf + row * LDSPITCH + (tid & 3) * 32;
    *(u32x4*)(p)      = reg[0];
    *(u32x4*)(p + 8)  = reg[1];
    *(u32x4*)(p + 16) = reg[2];
    *(u32x4*)(p + 24) = reg[3];
}
#endif

// ---------------------------------------------------------------------------
// One (m-superblock, n-tile) slot of one timestep.
//   slot = msb*64 + n_tile ; msb in {0,1} selects 256 batch rows.
//   wave wv owns rows m0 = msb*256 + wv*32 (two 16-row M-tiles).
// ---------------------------------------------------------------------------
__device__ __forceinline__ void
lstm_step_slot(const unsigned short* __restrict__ Xbf_t,   // [B, I]  bf16
               const unsigned short* __restrict__ Wihbf,   // [4H, I] bf16
               const unsigned short* __restrict__ Whhbf,   // [4H, H] bf16
               const float*          __restrict__ bias,    // [4H]
               const unsigned short* __restrict__ hbf_in,  // [B, H]  bf16
               float*                __restrict__ h,        // [B, H] fp32
               float*                __restrict__ c,        // [B, H] fp32
               unsigned short*       __restrict__ hbf_out,  // [B, H] bf16
               const float*          __restrict__ mask_t,   // [B]
               unsigned short*       __restrict__ lds0,
               unsigned short*       __restrict__ lds1,
               int slot, int tid)
{
    const int lane = tid & 31;
    const int wv   = tid >> 5;
    const int n0   = (slot & 63) << 4;
    const int msb  = slot >> 6;
    const int m0   = msb * 256 + wv * 32;               // first of two M-tiles
    const int col  = lane & 15;
    const int half = lane >> 4;

    // --- accumulators start at bias (broadcast across the 8 C rows) --------
    v8f acc0[4], acc1[4];
#pragma unroll
    for (int q = 0; q < 4; ++q) {
        const float bq = bias[q * HH + n0 + col];
#pragma unroll
        for (int j = 0; j < 8; ++j) { acc0[q][j] = bq; acc1[q][j] = bq; }
    }

    // --- per-lane A row bases (ISA 16-bit A fragment layout) ----------------
    const unsigned short* hA0 = hbf_in + (size_t)(m0 + col) * HH;
    const unsigned short* hA1 = hA0 + 16 * HH;

    // --- prologue: stage chunk 0 --------------------------------------------
#if HAVE_ASYNC_LDS
    stage_async(Whhbf, n0, 0, tid, lds0);
    __builtin_amdgcn_s_wait_asynccnt(0);
#else
    u32x4 stg[4];
    stage_fetch(Whhbf, n0, 0, tid, stg);
    stage_store(lds0, tid, stg);
#endif
    __syncthreads();

    // --- recurrent GEMM: K = 1024 in 8 LDS-staged chunks of 128 -------------
    for (int ch = 0; ch < HH / KC; ++ch) {
        const int kc = ch * KC;
        unsigned short* cur = (ch & 1) ? lds1 : lds0;
        unsigned short* nxt = (ch & 1) ? lds0 : lds1;
        if (ch < HH / KC - 1) {
#if HAVE_ASYNC_LDS
            stage_async(Whhbf, n0, kc + KC, tid, nxt);   // DMA overlaps WMMAs
#else
            stage_fetch(Whhbf, n0, kc + KC, tid, stg);   // overlap with WMMAs
#endif
        }
        const unsigned short* __restrict__ bufB = cur;
#pragma unroll
        for (int kk = 0; kk < KC; kk += 32) {
            const int kg = kc + kk;
            FragU a0, a1;
            a0.q[0] = *(const u32x4*)(hA0 + kg + 8 * half);
            a0.q[1] = *(const u32x4*)(hA0 + kg + 16 + 8 * half);
            a1.q[0] = *(const u32x4*)(hA1 + kg + 8 * half);
            a1.q[1] = *(const u32x4*)(hA1 + kg + 16 + 8 * half);
#pragma unroll
            for (int q = 0; q < 4; ++q) {
                FragU b;
                const unsigned short* bp =
                    bufB + (q * 16 + col) * LDSPITCH + kk + 8 * half;
                b.q[0] = *(const u32x4*)(bp);
                b.q[1] = *(const u32x4*)(bp + 16);
                acc0[q] = wmma_bf16(a0, b, acc0[q]);
                acc1[q] = wmma_bf16(a1, b, acc1[q]);
            }
        }
        if (ch < HH / KC - 1) {
#if HAVE_ASYNC_LDS
            __builtin_amdgcn_s_wait_asynccnt(0);
#else
            stage_store(nxt, tid, stg);
#endif
        }
        __syncthreads();
    }

    // --- input GEMM: K = 64 over x_t, direct global B ------------------------
    const unsigned short* xA0 = Xbf_t + (size_t)(m0 + col) * II;
    const unsigned short* xA1 = xA0 + 16 * II;
#pragma unroll
    for (int k0 = 0; k0 < II; k0 += 32) {
        FragU a0, a1;
        a0.q[0] = *(const u32x4*)(xA0 + k0 + 8 * half);
        a0.q[1] = *(const u32x4*)(xA0 + k0 + 16 + 8 * half);
        a1.q[0] = *(const u32x4*)(xA1 + k0 + 8 * half);
        a1.q[1] = *(const u32x4*)(xA1 + k0 + 16 + 8 * half);
#pragma unroll
        for (int q = 0; q < 4; ++q) {
            FragU b;
            const unsigned short* bp =
                Wihbf + (size_t)(q * HH + n0 + col) * II + k0 + 8 * half;
            b.q[0] = *(const u32x4*)(bp);
            b.q[1] = *(const u32x4*)(bp + 16);
            acc0[q] = wmma_bf16(a0, b, acc0[q]);
            acc1[q] = wmma_bf16(a1, b, acc1[q]);
        }
    }

    // --- pointwise LSTM cell update (fp32), mask blend, write-back ----------
    // C/D layout: VGPR r, lane -> element (M = r + 8*half, N = lane&15).
#pragma unroll
    for (int s = 0; s < 2; ++s) {
        const v8f* acc = (s == 0) ? acc0 : acc1;
        const int mbase = m0 + s * 16;
#pragma unroll
        for (int r = 0; r < 8; ++r) {
            const int brow = mbase + r + 8 * half;
            const int idx  = brow * HH + n0 + col;

            const float cold = c[idx];
            const float hold = h[idx];

            const float ig = sigmoid_f(acc[0][r]);
            const float fg = sigmoid_f(acc[1][r]);
            const float gg = tanh_f(acc[2][r]);
            const float og = sigmoid_f(acc[3][r]);

            const float cn = fg * cold + ig * gg;
            const float hn = og * tanh_f(cn);

            const float mv = mask_t[brow];
            const float hw = mv * hn + (1.0f - mv) * hold;
            const float cw = mv * cn + (1.0f - mv) * cold;

            h[idx]       = hw;
            c[idx]       = cw;
            hbf_out[idx] = f2bf_rne(hw);
        }
    }
}

// ---------------------------------------------------------------------------
// Persistent kernel: loops over all 128 timesteps; device-wide software
// barrier between steps (monotonic counter, no reset races).
// ---------------------------------------------------------------------------
__global__ void __launch_bounds__(256)
lstm_persistent_kernel(const unsigned short* __restrict__ Xbf,    // [T,B,I]
                       const unsigned short* __restrict__ Wihbf,
                       const unsigned short* __restrict__ Whhbf,
                       const float*          __restrict__ bias,
                       unsigned short*       __restrict__ hbf0,
                       unsigned short*       __restrict__ hbf1,
                       float*                __restrict__ h,
                       float*                __restrict__ c,
                       const float*          __restrict__ mask,   // [T,B]
                       unsigned int*         __restrict__ bar)
{
    __shared__ unsigned short ldsB[2][64 * LDSPITCH];   // 34 KB double buffer
    const int tid = threadIdx.x;

    for (int t = 0; t < TT; ++t) {
        const unsigned short* hin  = (t & 1) ? hbf1 : hbf0;
        unsigned short*       hout = (t & 1) ? hbf0 : hbf1;
        const unsigned short* Xt   = Xbf + (size_t)t * BB * II;
        const float*          mt   = mask + (size_t)t * BB;

        for (int slot = blockIdx.x; slot < NSLOTS; slot += gridDim.x) {
            lstm_step_slot(Xt, Wihbf, Whhbf, bias, hin, h, c, hout, mt,
                           ldsB[0], ldsB[1], slot, tid);
        }

        // ---- device-wide barrier: flush stores, arrive, spin, invalidate ----
        __builtin_amdgcn_fence(__ATOMIC_RELEASE, "agent");
        __syncthreads();
        if (tid == 0) {
            __hip_atomic_fetch_add(bar, 1u, __ATOMIC_RELAXED,
                                   __HIP_MEMORY_SCOPE_AGENT);
            const unsigned int target = (unsigned int)gridDim.x * (unsigned int)(t + 1);
            while (__hip_atomic_load(bar, __ATOMIC_RELAXED,
                                     __HIP_MEMORY_SCOPE_AGENT) < target) {
                __builtin_amdgcn_s_sleep(2);
            }
        }
        __syncthreads();
        __builtin_amdgcn_fence(__ATOMIC_ACQUIRE, "agent");
    }
}

// ---------------------------------------------------------------------------
// Final output: d_out = [h (B*H) | c (B*H)] fp32.
// ---------------------------------------------------------------------------
__global__ void lstm_copyout_kernel(const float* __restrict__ h,
                                    const float* __restrict__ c,
                                    float* __restrict__ out) {
    const unsigned int i = blockIdx.x * blockDim.x + threadIdx.x;  // 524,288
    out[i]           = h[i];
    out[BB * HH + i] = c[i];
}

// ---------------------------------------------------------------------------

extern "C" void kernel_launch(void* const* d_in, const int* in_sizes, int n_in,
                              void* d_out, int out_size, void* d_ws, size_t ws_size,
                              hipStream_t stream) {
    const float* X    = (const float*)d_in[0];   // [T, B, I]
    const float* mask = (const float*)d_in[1];   // [T, 1, B]
    const float* Wih  = (const float*)d_in[2];   // [4H, I]
    const float* Whh  = (const float*)d_in[3];   // [4H, H]
    const float* bih  = (const float*)d_in[4];   // [4H]
    const float* bhh  = (const float*)d_in[5];   // [4H]
    const float* h0   = (const float*)d_in[6];   // [1, H]
    const float* c0   = (const float*)d_in[7];   // [1, H]
    float* out = (float*)d_out;

    // Workspace layout (all segments 256B-aligned; total ~22.6 MB)
    char* ws = (char*)d_ws;
    unsigned short* Wihbf = (unsigned short*)(ws + 0x0000000);  // 512 KB
    unsigned short* Whhbf = (unsigned short*)(ws + 0x0080000);  // 8 MB
    float*          bias  = (float*)         (ws + 0x0880000);  // 16 KB
    unsigned short* Xbf   = (unsigned short*)(ws + 0x0884000);  // 8 MB
    float*          hbuf  = (float*)         (ws + 0x1084000);  // 2 MB
    float*          cbuf  = (float*)         (ws + 0x1284000);  // 2 MB
    unsigned short* hbf0  = (unsigned short*)(ws + 0x1484000);  // 1 MB
    unsigned short* hbf1  = (unsigned short*)(ws + 0x1584000);  // 1 MB
    unsigned int*   bar   = (unsigned int*)  (ws + 0x1684000);  // 4 B

    // Grid for the persistent kernel: never exceed the processor count so all
    // blocks are co-resident (deadlock-free software barrier). Deterministic
    // per device.
    int dev = 0;
    (void)hipGetDevice(&dev);
    int mp = NSLOTS;
    (void)hipDeviceGetAttribute(&mp, hipDeviceAttributeMultiprocessorCount, dev);
    const int nblocks = (mp < 1) ? 1 : ((mp > NSLOTS) ? NSLOTS : mp);

    // 1) Pack weights/inputs to bf16, fuse biases, init state, reset barrier.
    lstm_pack_kernel<<<dim3(16384), dim3(256), 0, stream>>>(
        X, Wih, Whh, bih, bhh, h0, c0,
        Xbf, Wihbf, Whhbf, bias, hbuf, cbuf, hbf0, bar);

    // 2) All 128 timesteps in one persistent kernel.
    lstm_persistent_kernel<<<dim3(nblocks), dim3(256), 0, stream>>>(
        Xbf, Wihbf, Whhbf, bias, hbf0, hbf1, hbuf, cbuf, mask, bar);

    // 3) Emit (h, c) concatenated.
    lstm_copyout_kernel<<<dim3(2048), dim3(256), 0, stream>>>(hbuf, cbuf, out);
}
